// FasterRCNN_45286135169232
// MI455X (gfx1250) — compile-verified, matrix operand order checked by reading, not run on hardware
//
#include <hip/hip_runtime.h>
#include <hip/hip_bf16.h>

// ---------------------------------------------------------------------------
// Types for CDNA5 WMMA
// ---------------------------------------------------------------------------
typedef __attribute__((ext_vector_type(16))) __bf16        v16bf;
typedef __attribute__((ext_vector_type(8)))  float         v8f;
typedef __attribute__((ext_vector_type(4)))  float         f32x4;
typedef __attribute__((ext_vector_type(4)))  unsigned int  u32x4;

union Frag16 { v16bf v; u32x4 q[2]; };

// Pack two f32 into two bf16 (round-half-up) with a single v_perm_b32:
// result = [hi16(lo+0x8000) , hi16(hi+0x8000)]
__device__ __forceinline__ unsigned int bfpack2(float lo, float hi) {
  unsigned int a = __float_as_uint(lo) + 0x8000u;
  unsigned int b = __float_as_uint(hi) + 0x8000u;
  return __builtin_amdgcn_perm(b, a, 0x07060302u);  // bytes: [a2,a3,b2,b3]
}
__device__ __forceinline__ unsigned short f2bf(float f) {
  unsigned int u = __float_as_uint(f);
  u += 0x7FFFu + ((u >> 16) & 1u);
  return (unsigned short)(u >> 16);
}
__device__ __forceinline__ float bf2f(unsigned short u) {
  return __uint_as_float(((unsigned int)u) << 16);
}

// ---------------------------------------------------------------------------
// Fused GEMM: out = relu(A @ B^T + bias), bf16 out, f32 WMMA accumulation.
// A: [M,K] (f32 if A_F32 else bf16)   B: [Nn,K] f32 (bf16-converted on load)
// 128x128 workgroup tile, BK=32, 8 waves (4 M x 2 N), 2x4 WMMA accs per wave.
// Double-buffered LDS; GEMM2's bf16 A-tile uses global_load_async_to_lds_b128.
// ---------------------------------------------------------------------------
template <bool A_F32>
__global__ __launch_bounds__(256) void gemm_bias_relu_bf16(
    const void* __restrict__ Av, const float* __restrict__ B,
    const float* __restrict__ bias, unsigned short* __restrict__ out,
    int M, int Nn, int K)
{
  constexpr int BM = 128, BN = 128, BK = 32, STR = BK + 8;  // 80B LDS row
  __shared__ __align__(16) unsigned short As[2][BM * STR];
  __shared__ __align__(16) unsigned short Bs[2][BN * STR];

  const int tid   = threadIdx.x;
  const int lane  = tid & 31, wid = tid >> 5;
  const int waveM = wid & 3,  waveN = wid >> 2;
  const int hi    = lane >> 4, lm = lane & 15;

  const int n0 = blockIdx.x * BN;   // x = N-tile (fast axis -> A panel L2 reuse)
  const int m0 = blockIdx.y * BM;   // y = M-tile

  // cooperative loader coords: 256 threads, each moves 16 elems of A and B
  const int lrow = tid >> 1;         // 0..127
  const int lk0  = (tid & 1) * 16;   // 0 or 16
  const int aRow = min(m0 + lrow, M - 1);
  const int bRow = min(n0 + lrow, Nn - 1);

  const float*          Af = (const float*)Av;
  const unsigned short* Ab = (const unsigned short*)Av;

  f32x4 ar[4];   // staged A (f32 path)
  f32x4 br_[4];  // staged B

  // Issue global loads for K-panel kt; bf16 A goes straight to LDS via async.
  auto issueLoads = [&](int kt, int buf) {
    if (A_F32) {
      const f32x4* gp = (const f32x4*)(Af + (size_t)aRow * K + kt + lk0);
      ar[0] = gp[0]; ar[1] = gp[1]; ar[2] = gp[2]; ar[3] = gp[3];
    } else {
      unsigned int lds0 = (unsigned int)(size_t)&As[buf][lrow * STR + lk0];
      unsigned long long ga =
          (unsigned long long)(Ab + (size_t)aRow * K + kt + lk0);
      asm volatile(
          "global_load_async_to_lds_b128 %0, %1, off\n\t"
          "global_load_async_to_lds_b128 %0, %1, off offset:16"
          :: "v"(lds0), "v"(ga) : "memory");
    }
    const f32x4* gp = (const f32x4*)(B + (size_t)bRow * K + kt + lk0);
    br_[0] = gp[0]; br_[1] = gp[1]; br_[2] = gp[2]; br_[3] = gp[3];
  };

  // Convert staged registers to bf16 and store to LDS buffer `buf`.
  auto commitLds = [&](int buf) {
    if (A_F32) {
      u32x4 q0 = { bfpack2(ar[0].x, ar[0].y), bfpack2(ar[0].z, ar[0].w),
                   bfpack2(ar[1].x, ar[1].y), bfpack2(ar[1].z, ar[1].w) };
      u32x4 q1 = { bfpack2(ar[2].x, ar[2].y), bfpack2(ar[2].z, ar[2].w),
                   bfpack2(ar[3].x, ar[3].y), bfpack2(ar[3].z, ar[3].w) };
      *(u32x4*)&As[buf][lrow * STR + lk0    ] = q0;
      *(u32x4*)&As[buf][lrow * STR + lk0 + 8] = q1;
    }
    u32x4 q0 = { bfpack2(br_[0].x, br_[0].y), bfpack2(br_[0].z, br_[0].w),
                 bfpack2(br_[1].x, br_[1].y), bfpack2(br_[1].z, br_[1].w) };
    u32x4 q1 = { bfpack2(br_[2].x, br_[2].y), bfpack2(br_[2].z, br_[2].w),
                 bfpack2(br_[3].x, br_[3].y), bfpack2(br_[3].z, br_[3].w) };
    *(u32x4*)&Bs[buf][lrow * STR + lk0    ] = q0;
    *(u32x4*)&Bs[buf][lrow * STR + lk0 + 8] = q1;
  };

  v8f acc[2][4] = {};

  // ---- prologue: fill buffer 0 ----
  issueLoads(0, 0);
  commitLds(0);
  if (!A_F32) asm volatile("s_wait_asynccnt 0x0" ::: "memory");
  __syncthreads();

  int p = 0;
  for (int kt = 0; kt < K; kt += BK) {
    const bool hasNext = (kt + BK) < K;
    if (hasNext) issueLoads(kt + BK, p ^ 1);  // overlap with WMMA below

    // ---- fragments: two ds_load_b128 per 16x32 bf16 tile ----
    // 16-bit A layout: lane hi=0 -> K 0..7 & 16..23, hi=1 -> K 8..15 & 24..31
    Frag16 a[2], b[4];
    #pragma unroll
    for (int im = 0; im < 2; ++im) {
      int base = (waveM * 32 + im * 16 + lm) * STR;
      a[im].q[0] = *(const u32x4*)&As[p][base + hi * 8];
      a[im].q[1] = *(const u32x4*)&As[p][base + 16 + hi * 8];
    }
    #pragma unroll
    for (int in = 0; in < 4; ++in) {
      int base = (waveN * 64 + in * 16 + lm) * STR;
      b[in].q[0] = *(const u32x4*)&Bs[p][base + hi * 8];
      b[in].q[1] = *(const u32x4*)&Bs[p][base + 16 + hi * 8];
    }

    #pragma unroll
    for (int im = 0; im < 2; ++im)
      #pragma unroll
      for (int in = 0; in < 4; ++in)
        acc[im][in] = __builtin_amdgcn_wmma_f32_16x16x32_bf16(
            false, a[im].v, false, b[in].v, (short)0, acc[im][in], false, false);

    if (hasNext) commitLds(p ^ 1);
    if (!A_F32) asm volatile("s_wait_asynccnt 0x0" ::: "memory");
    __syncthreads();
    p ^= 1;
  }

  // ---- epilogue: bias + ReLU + bf16 store (C/D layout: VGPR e -> M=e+hi*8) --
  #pragma unroll
  for (int in = 0; in < 4; ++in) {
    int col = n0 + waveN * 64 + in * 16 + lm;
    float bv = bias[col];
    #pragma unroll
    for (int im = 0; im < 2; ++im) {
      int rowBase = m0 + waveM * 32 + im * 16 + hi * 8;
      #pragma unroll
      for (int e = 0; e < 8; ++e) {
        int row = rowBase + e;
        if (row < M) {
          float x = acc[im][in][e] + bv;
          x = x > 0.f ? x : 0.f;
          out[(size_t)row * Nn + col] = f2bf(x);
        }
      }
    }
  }
}

// ---------------------------------------------------------------------------
// Predictor head + box decode: one wave32 per ROI.
// ---------------------------------------------------------------------------
__global__ __launch_bounds__(256) void head_decode_kernel(
    const unsigned short* __restrict__ v2, const float* __restrict__ rois,
    const float* __restrict__ wc, const float* __restrict__ bc,
    const float* __restrict__ wr, const float* __restrict__ br,
    float* __restrict__ oScores, float* __restrict__ oClasses,
    float* __restrict__ oBoxes, float* __restrict__ mscore, int M, int K)
{
  const int lane = threadIdx.x & 31, wid = threadIdx.x >> 5;
  const int row = blockIdx.x * 8 + wid;
  if (row >= M) return;

  float c[4]  = {};
  float r[16] = {};
  const unsigned short* vr = v2 + (size_t)row * K;
  for (int k = lane; k < K; k += 32) {
    float x = bf2f(vr[k]);
    #pragma unroll
    for (int j = 0; j < 4;  ++j) c[j] += x * wc[j * K + k];
    #pragma unroll
    for (int j = 0; j < 16; ++j) r[j] += x * wr[j * K + k];
  }
  #pragma unroll
  for (int off = 16; off > 0; off >>= 1) {
    #pragma unroll
    for (int j = 0; j < 4;  ++j) c[j] += __shfl_xor(c[j], off, 32);
    #pragma unroll
    for (int j = 0; j < 16; ++j) r[j] += __shfl_xor(r[j], off, 32);
  }
  if (lane == 0) {
    int   cls   = 0;
    float score = c[0] + bc[0];
    #pragma unroll
    for (int j = 1; j < 4; ++j) {
      float cj = c[j] + bc[j];
      if (cj > score) { score = cj; cls = j; }
    }
    float d0 = r[cls * 4 + 0] + br[cls * 4 + 0];
    float d1 = r[cls * 4 + 1] + br[cls * 4 + 1];
    float d2 = r[cls * 4 + 2] + br[cls * 4 + 2];
    float d3 = r[cls * 4 + 3] + br[cls * 4 + 3];

    float x1 = rois[row * 4 + 0], y1 = rois[row * 4 + 1];
    float x2 = rois[row * 4 + 2], y2 = rois[row * 4 + 3];
    float w  = x2 - x1, h = y2 - y1;
    float cx = x1 + 0.5f * w, cy = y1 + 0.5f * h;
    float pcx = cx + d0 * 0.1f * w;
    float pcy = cy + d1 * 0.1f * h;
    float pw  = __expf(d2 * 0.2f) * w;
    float ph  = __expf(d3 * 0.2f) * h;
    float bx1 = fminf(fmaxf(pcx - 0.5f * pw, 0.f), 512.f);
    float by1 = fminf(fmaxf(pcy - 0.5f * ph, 0.f), 512.f);
    float bx2 = fminf(fmaxf(pcx + 0.5f * pw, 0.f), 512.f);
    float by2 = fminf(fmaxf(pcy + 0.5f * ph, 0.f), 512.f);

    oScores[row]  = score;
    oClasses[row] = (float)cls;
    oBoxes[row * 4 + 0] = bx1;
    oBoxes[row * 4 + 1] = by1;
    oBoxes[row * 4 + 2] = bx2;
    oBoxes[row * 4 + 3] = by2;
    mscore[row] = (cls != 3) ? score : -__builtin_inff();
  }
}

// ---------------------------------------------------------------------------
// Bitonic sort (descending by masked score), single block, 8192 slots in LDS.
// ---------------------------------------------------------------------------
__global__ __launch_bounds__(1024) void nms_sort_kernel(
    const float* __restrict__ mscore, const float* __restrict__ boxes,
    int N, int* __restrict__ sidx, float* __restrict__ sboxes)
{
  constexpr int SN = 8192;
  __shared__ float key[SN];
  __shared__ int   idx[SN];
  const int tid = threadIdx.x;
  for (int i = tid; i < SN; i += 1024) {
    key[i] = (i < N) ? mscore[i] : -__builtin_inff();
    idx[i] = i;
  }
  __syncthreads();
  for (int k = 2; k <= SN; k <<= 1) {
    for (int j = k >> 1; j > 0; j >>= 1) {
      for (int t = tid; t < SN / 2; t += 1024) {
        int a = ((t / j) * (j << 1)) + (t % j);
        int b = a + j;
        bool desc = ((a & k) == 0);
        float ka = key[a], kb = key[b];
        bool sw = desc ? (ka < kb) : (ka > kb);
        if (sw) {
          key[a] = kb; key[b] = ka;
          int ti = idx[a]; idx[a] = idx[b]; idx[b] = ti;
        }
      }
      __syncthreads();
    }
  }
  for (int i = tid; i < N; i += 1024) {
    int o = idx[i];
    sidx[i] = o;
    sboxes[i * 4 + 0] = boxes[o * 4 + 0];
    sboxes[i * 4 + 1] = boxes[o * 4 + 1];
    sboxes[i * 4 + 2] = boxes[o * 4 + 2];
    sboxes[i * 4 + 3] = boxes[o * 4 + 3];
  }
}

// ---------------------------------------------------------------------------
// Suppression bitmask over sorted order (torchvision-style), wave32 blocks.
// ---------------------------------------------------------------------------
__global__ __launch_bounds__(32) void nms_mask_kernel(
    const float* __restrict__ sboxes, unsigned int* __restrict__ mask,
    int N, int nb)
{
  const int bi = blockIdx.y, bj = blockIdx.x;
  const int i  = bi * 32 + threadIdx.x;
  const int ic = min(i, N - 1);
  const float ax1 = sboxes[ic * 4 + 0], ay1 = sboxes[ic * 4 + 1];
  const float ax2 = sboxes[ic * 4 + 2], ay2 = sboxes[ic * 4 + 3];
  const float areaA = (ax2 - ax1) * (ay2 - ay1);
  unsigned int bits = 0u;
  if (bj >= bi) {
    for (int jj = 0; jj < 32; ++jj) {
      int j = bj * 32 + jj;
      if (j < N && j > i) {
        float bx1 = sboxes[j * 4 + 0], by1 = sboxes[j * 4 + 1];
        float bx2 = sboxes[j * 4 + 2], by2 = sboxes[j * 4 + 3];
        float ix = fminf(ax2, bx2) - fmaxf(ax1, bx1);
        float iy = fminf(ay2, by2) - fmaxf(ay1, by1);
        float inter = fmaxf(ix, 0.f) * fmaxf(iy, 0.f);
        float areaB = (bx2 - bx1) * (by2 - by1);
        float iou = inter / (areaA + areaB - inter + 1e-8f);
        if (iou > 0.2f) bits |= (1u << jj);
      }
    }
  }
  if (i < N) mask[(size_t)i * nb + bj] = bits;
}

// ---------------------------------------------------------------------------
// Sequential greedy sweep over the bitmask; scatter keep to original order.
// ---------------------------------------------------------------------------
__global__ __launch_bounds__(256) void nms_reduce_kernel(
    const int* __restrict__ sidx, const unsigned int* __restrict__ mask,
    const float* __restrict__ oClasses, float* __restrict__ oKeep,
    int N, int nb)
{
  __shared__ unsigned int remv[256];
  __shared__ int curKeep;
  const int tid = threadIdx.x;
  for (int t = tid; t < nb; t += 256) remv[t] = 0u;
  __syncthreads();
  for (int i = 0; i < N; ++i) {
    if (tid == 0) curKeep = !((remv[i >> 5] >> (i & 31)) & 1u);
    __syncthreads();
    if (curKeep) {
      for (int t = tid; t < nb; t += 256)
        remv[t] |= mask[(size_t)i * nb + t];
    }
    if (tid == 0) {
      int orig = sidx[i];
      bool valid = oClasses[orig] != 3.0f;
      oKeep[orig] = (curKeep && valid) ? 1.0f : 0.0f;
    }
    __syncthreads();
  }
}

// ---------------------------------------------------------------------------
// Launch
// ---------------------------------------------------------------------------
extern "C" void kernel_launch(void* const* d_in, const int* in_sizes, int n_in,
                              void* d_out, int out_size, void* d_ws, size_t ws_size,
                              hipStream_t stream)
{
  const float* f    = (const float*)d_in[0];
  const float* rois = (const float*)d_in[1];
  const float* w6   = (const float*)d_in[2];
  const float* b6   = (const float*)d_in[3];
  const float* w7   = (const float*)d_in[4];
  const float* b7   = (const float*)d_in[5];
  const float* wc   = (const float*)d_in[6];
  const float* bc   = (const float*)d_in[7];
  const float* wr   = (const float*)d_in[8];
  const float* br   = (const float*)d_in[9];

  const int N = in_sizes[1] / 4;   // 8000 rois
  const int H = in_sizes[3];       // 1024
  const int D = in_sizes[0] / N;   // 12544

  float* out      = (float*)d_out;
  float* oScores  = out;
  float* oClasses = out + N;
  float* oBoxes   = out + 2 * N;
  float* oKeep    = out + 6 * N;

  char* ws = (char*)d_ws;
  size_t off = 0;
  auto carve = [&](size_t bytes) -> void* {
    void* p = ws + off;
    off = (off + bytes + 255) & ~(size_t)255;
    return p;
  };
  unsigned short* v1 = (unsigned short*)carve((size_t)N * H * 2);
  unsigned short* v2 = (unsigned short*)carve((size_t)N * H * 2);
  float* mscore      = (float*)carve((size_t)N * 4);
  int*   sidx        = (int*)carve((size_t)8192 * 4);
  float* sboxes      = (float*)carve((size_t)N * 4 * 4);
  const int nb       = (N + 31) / 32;
  unsigned int* mask = (unsigned int*)carve((size_t)N * nb * 4);
  (void)ws_size; (void)n_in; (void)out_size;

  // x = N-tiles (fast), y = M-tiles: consecutive blocks share the A row panel
  dim3 g1(H / 128, (N + 127) / 128);
  gemm_bias_relu_bf16<true ><<<g1, 256, 0, stream>>>(f,  w6, b6, v1, N, H, D);
  gemm_bias_relu_bf16<false><<<g1, 256, 0, stream>>>(v1, w7, b7, v2, N, H, H);
  head_decode_kernel<<<(N + 7) / 8, 256, 0, stream>>>(
      v2, rois, wc, bc, wr, br, oScores, oClasses, oBoxes, mscore, N, H);
  nms_sort_kernel<<<1, 1024, 0, stream>>>(mscore, oBoxes, N, sidx, sboxes);
  nms_mask_kernel<<<dim3(nb, nb), 32, 0, stream>>>(sboxes, mask, N, nb);
  nms_reduce_kernel<<<1, 256, 0, stream>>>(sidx, mask, oClasses, oKeep, N, nb);
}
